// STraTS_MultiTimeAttention_65034394796063
// MI455X (gfx1250) — matrix operand order repaired and checked by reading, not compiled, not run on hardware
//
#include <hip/hip_runtime.h>

#define MASKV (-1e-30f)

typedef __bf16 bf16_t;
typedef __attribute__((ext_vector_type(16))) __bf16 v16bf;
typedef __attribute__((ext_vector_type(8)))  __bf16 bf16x8;
typedef __attribute__((ext_vector_type(8)))  float  v8f;
typedef __attribute__((ext_vector_type(4)))  float  f32x4;
typedef __attribute__((ext_vector_type(4)))  unsigned int u32x4;
typedef __attribute__((ext_vector_type(8)))  int  i32x8;
typedef __attribute__((ext_vector_type(4)))  int  i32x4;

#if __has_builtin(__builtin_amdgcn_tensor_load_to_lds)
#define HAVE_TDM 1
#else
#define HAVE_TDM 0
#endif

__device__ __forceinline__ v16bf cat16(bf16x8 lo, bf16x8 hi) {
  v16bf r;
#pragma unroll
  for (int i = 0; i < 8; ++i) { r[i] = lo[i]; r[8 + i] = hi[i]; }
  return r;
}

// B-fragment: 16 contiguous K values per lane (lanes 0-15: K 0..15, lanes 16-31: K 16..31)
__device__ __forceinline__ v16bf ldB(const bf16_t* p) {
  return cat16(*(const bf16x8*)p, *(const bf16x8*)(p + 8));
}

__device__ __forceinline__ v8f wmma_bf16(v16bf a, v16bf b, v8f c) {
  return __builtin_amdgcn_wmma_f32_16x16x32_bf16(
      false, a, false, b, (short)0, c, false, false);
}

__device__ __forceinline__ float hmax16(float v) {
  v = fmaxf(v, __shfl_xor(v, 1, 32));
  v = fmaxf(v, __shfl_xor(v, 2, 32));
  v = fmaxf(v, __shfl_xor(v, 4, 32));
  v = fmaxf(v, __shfl_xor(v, 8, 32));
  return v;
}
__device__ __forceinline__ float hsum16(float v) {
  v += __shfl_xor(v, 1, 32);
  v += __shfl_xor(v, 2, 32);
  v += __shfl_xor(v, 4, 32);
  v += __shfl_xor(v, 8, 32);
  return v;
}

#if HAVE_TDM
// TDM: DMA a 64-row x 32-col bf16 tile (row stride = stride_elems) from global
// into LDS at lds_addr, laid out contiguously [64][32].  D# per CDNA5 ISA §8.
__device__ __forceinline__ void tdm_load_tile_32x64(unsigned lds_addr,
                                                    const bf16_t* gp,
                                                    unsigned stride_elems) {
  unsigned long long ga = (unsigned long long)(uintptr_t)gp;
  u32x4 g0;
  g0[0] = 1u;                                     // count=1, user mode
  g0[1] = lds_addr;                               // lds_addr (bytes)
  g0[2] = (unsigned)ga;                           // global_addr[31:0]
  g0[3] = (unsigned)((ga >> 32) & 0x01ffffffu)    // global_addr[56:32]
          | 0x80000000u;                          // type=2 ("image")
  i32x8 g1;
  g1[0] = 0x00010000;                             // wg_mask=0, data_size=2B
  g1[1] = (int)((stride_elems & 0xffffu) << 16);  // tensor_dim0[15:0]
  g1[2] = (int)((stride_elems >> 16) | (64u << 16));  // dim0[31:16] | dim1[15:0]=64
  g1[3] = (int)(32u << 16);                       // tile_dim0 = 32
  g1[4] = 64;                                     // tile_dim1 = 64
  g1[5] = (int)stride_elems;                      // tensor_dim0_stride[31:0]
  g1[6] = 0;
  g1[7] = 0;
  i32x4 z4 = {0, 0, 0, 0};
#if __clang_major__ >= 23
  i32x8 z8 = {0, 0, 0, 0, 0, 0, 0, 0};
  __builtin_amdgcn_tensor_load_to_lds(g0, g1, z4, z4, z8, 0);
#else
  __builtin_amdgcn_tensor_load_to_lds(g0, g1, z4, z4, 0);
#endif
}
#endif

// cooperative fallback staging (256 threads): copy 64x32 bf16 tile to LDS
__device__ __forceinline__ void stage_tile_32x64(bf16_t* dst, const bf16_t* gp,
                                                 unsigned stride_elems) {
  const int id = threadIdx.x * 8;          // 2048 elems / 256 threads
  const int c = id >> 5, kc = id & 31;
  *(bf16x8*)(dst + id) = *(const bf16x8*)(gp + (size_t)c * stride_elems + kc);
}

// ---------------------------------------------------------------------------
// Kernel 0: one-shot weight prep: f32 -> bf16, transposed to [col][k].
//   Wt  : [which][h][c(64)][k(512)]   (3*8*64*512 elems)
//   WoT : [c(512)][k(512)]            (512*512 elems)
// ---------------------------------------------------------------------------
__global__ __launch_bounds__(256) void prep_weights(
    const float* __restrict__ Wq, const float* __restrict__ Wk,
    const float* __restrict__ Wv, const float* __restrict__ Wo,
    bf16_t* __restrict__ Wt, bf16_t* __restrict__ WoT) {
  const int t = blockIdx.x * 256 + threadIdx.x;
  if (t < 3 * 8 * 64 * 512) {
    const int which = t / (8 * 64 * 512);
    const int r = t % (8 * 64 * 512);
    const int hh = r / (64 * 512);
    const int c = (r / 512) & 63;
    const int kk = r & 511;
    const float* W = (which == 0 ? Wq : which == 1 ? Wk : Wv);
    Wt[t] = (bf16_t)W[((size_t)hh * 512 + kk) * 64 + c];
  } else {
    const int u = t - 3 * 8 * 64 * 512;     // 0 .. 512*512-1
    const int c = u >> 9, kk = u & 511;
    WoT[u] = (bf16_t)Wo[((size_t)kk << 9) + c];
  }
}

// ---------------------------------------------------------------------------
// Kernel 1: Q/K/V projections.  X(512x512 f32) * W -> bf16.
// grid (S/128, B*h, 3), block 256 (8 waves); TDM double-buffers W tiles.
// ---------------------------------------------------------------------------
__global__ __launch_bounds__(256) void proj_kernel(
    const float* __restrict__ q, const float* __restrict__ k,
    const float* __restrict__ v, const bf16_t* __restrict__ Wt,
    bf16_t* __restrict__ Qb, bf16_t* __restrict__ Kb, bf16_t* __restrict__ Vt) {
  constexpr int S = 512, D = 512, DK = 64;
  const int bh = blockIdx.y;
  const int b = bh >> 3;
  const int which = blockIdx.z;
  const float* X = (which == 0 ? q : which == 1 ? k : v) + (size_t)b * S * D;
  const bf16_t* W = Wt + ((size_t)which * 8 + (bh & 7)) * 64 * 512;  // [c][k]

  const int wave = threadIdx.x >> 5, lane = threadIdx.x & 31;
  const int lo = lane & 15, hi = lane >> 4;
  const int koff = hi * 8, hi16 = hi * 16;
  const int wrow = blockIdx.x * 128 + wave * 16;

  __shared__ __align__(16) bf16_t wbuf[2][64 * 32];   // [c][kk] per buffer

  const float* xp = X + (size_t)(wrow + lo) * D;
  v8f acc[4] = {};

#if HAVE_TDM
  if (wave == 0) tdm_load_tile_32x64((unsigned)(uintptr_t)&wbuf[0][0], W, 512);
#else
  stage_tile_32x64(wbuf[0], W, 512);
#endif

  int p = 0;
#pragma unroll 1
  for (int kk = 0; kk < D; kk += 32) {
    __syncthreads();      // nobody still reads wbuf[p^1]
#if HAVE_TDM
    if (wave == 0) {
      if (kk + 32 < D)
        tdm_load_tile_32x64((unsigned)(uintptr_t)&wbuf[p ^ 1][0], W + kk + 32, 512);
      if (kk + 32 < D) __builtin_amdgcn_s_wait_tensorcnt((short)1);
      else             __builtin_amdgcn_s_wait_tensorcnt((short)0);
    }
#else
    if (kk + 32 < D) stage_tile_32x64(wbuf[p ^ 1], W + kk + 32, 512);
#endif
    __syncthreads();      // wbuf[p] ready for everyone

    __builtin_prefetch(xp + kk + 64, 0, 0);

    // A fragment (f32 -> bf16 on the fly)
    f32x4 x0 = *(const f32x4*)(xp + kk + hi16);
    f32x4 x1 = *(const f32x4*)(xp + kk + hi16 + 4);
    f32x4 x2 = *(const f32x4*)(xp + kk + hi16 + 8);
    f32x4 x3 = *(const f32x4*)(xp + kk + hi16 + 12);
    v16bf a;
#pragma unroll
    for (int i = 0; i < 4; ++i) {
      a[i] = (bf16_t)x0[i];      a[4 + i]  = (bf16_t)x1[i];
      a[8 + i] = (bf16_t)x2[i];  a[12 + i] = (bf16_t)x3[i];
    }

    // preload all B fragments, then back-to-back WMMAs
    const bf16_t* wb = wbuf[p];
    v16bf b0 = ldB(wb + (0 * 16 + lo) * 32 + hi16);
    v16bf b1 = ldB(wb + (1 * 16 + lo) * 32 + hi16);
    v16bf b2 = ldB(wb + (2 * 16 + lo) * 32 + hi16);
    v16bf b3 = ldB(wb + (3 * 16 + lo) * 32 + hi16);
    acc[0] = wmma_bf16(a, b0, acc[0]);
    acc[1] = wmma_bf16(a, b1, acc[1]);
    acc[2] = wmma_bf16(a, b2, acc[2]);
    acc[3] = wmma_bf16(a, b3, acc[3]);
    p ^= 1;
  }

  const size_t bhS = (size_t)bh * S;
  if (which < 2) {                         // Q, K row-major [t][dk]
    bf16_t* out = (which == 0 ? Qb : Kb);
#pragma unroll
    for (int g = 0; g < 4; ++g)
#pragma unroll
      for (int j = 0; j < 8; ++j)
        out[(bhS + wrow + j + koff) * DK + g * 16 + lo] = (bf16_t)acc[g][j];
  } else {                                 // V transposed [c][t] -> contiguous b128
#pragma unroll
    for (int g = 0; g < 4; ++g) {
      bf16x8 pk;
#pragma unroll
      for (int j = 0; j < 8; ++j) pk[j] = (bf16_t)acc[g][j];
      *(bf16x8*)(Vt + ((size_t)bh * DK + g * 16 + lo) * S + wrow + koff) = pk;
    }
  }
}

// ---------------------------------------------------------------------------
// Kernel 2: flash attention per (b,h).  grid (S/64, B*h), block 128 (4 waves);
// wave owns a 16-query tile, online-softmax over key tiles of 32.
// ---------------------------------------------------------------------------
__global__ __launch_bounds__(128) void attn_kernel(
    const bf16_t* __restrict__ Qb, const bf16_t* __restrict__ Kb,
    const bf16_t* __restrict__ Vt, const float* __restrict__ mask,
    bf16_t* __restrict__ Oc) {
  constexpr int S = 512, DK = 64, HD = 512;
  const int bh = blockIdx.y;
  const int b = bh >> 3, hh = bh & 7;
  const int wave = threadIdx.x >> 5, lane = threadIdx.x & 31;
  const int lo = lane & 15, hi = lane >> 4;
  const int koff = hi * 8, hi16 = hi * 16;
  const int s0 = blockIdx.x * 64 + wave * 16;

  const bf16_t* Q = Qb + (size_t)bh * S * DK;
  const bf16_t* K = Kb + (size_t)bh * S * DK;
  const bf16_t* V = Vt + (size_t)bh * DK * S;
  const float* mp = mask + (size_t)b * S;

  __shared__ __align__(16) bf16_t pst[4][16 * 32];
  bf16_t* ps = pst[wave];

  const bf16_t* qp = Q + (size_t)(s0 + lo) * DK;
  v16bf aQ0 = cat16(*(const bf16x8*)(qp + hi16),      *(const bf16x8*)(qp + 8 + hi16));
  v16bf aQ1 = cat16(*(const bf16x8*)(qp + 32 + hi16), *(const bf16x8*)(qp + 40 + hi16));

  float m[8], l[8];
  v8f o[4] = {};
#pragma unroll
  for (int j = 0; j < 8; ++j) { m[j] = -1e30f; l[j] = 0.f; }

#pragma unroll 1
  for (int t0 = 0; t0 < S; t0 += 32) {
    // --- scores: preload 4 K fragments, interleaved WMMAs
    const bf16_t* kp0 = K + (size_t)(t0 + lo) * DK;
    const bf16_t* kp1 = kp0 + 16 * DK;
    v16bf k00 = ldB(kp0 + hi16);
    v16bf k01 = ldB(kp0 + 32 + hi16);
    v16bf k10 = ldB(kp1 + hi16);
    v16bf k11 = ldB(kp1 + 32 + hi16);
    v8f C0 = {}, C1 = {};
    C0 = wmma_bf16(aQ0, k00, C0);
    C1 = wmma_bf16(aQ0, k10, C1);
    C0 = wmma_bf16(aQ1, k01, C0);
    C1 = wmma_bf16(aQ1, k11, C1);

    // --- mask:  A = m*A + (1-m)*MASKV  (column = lane)
    const float mv0 = mp[t0 + lo], mv1 = mp[t0 + 16 + lo];
#pragma unroll
    for (int j = 0; j < 8; ++j) {
      C0[j] = mv0 * C0[j] + (1.f - mv0) * MASKV;
      C1[j] = mv1 * C1[j] + (1.f - mv1) * MASKV;
    }

    // --- online softmax; stage P into LDS for layout swap (C -> A)
#pragma unroll
    for (int j = 0; j < 8; ++j) {
      float rm = hmax16(fmaxf(C0[j], C1[j]));
      float mn = fmaxf(m[j], rm);
      float sc = __expf(m[j] - mn);
      float p0 = __expf(C0[j] - mn);
      float p1 = __expf(C1[j] - mn);
      l[j] = l[j] * sc + hsum16(p0 + p1);
      m[j] = mn;
      o[0][j] *= sc; o[1][j] *= sc; o[2][j] *= sc; o[3][j] *= sc;
      const int M = j + koff;
      ps[M * 32 + lo]      = (bf16_t)p0;
      ps[M * 32 + 16 + lo] = (bf16_t)p1;
    }
    __builtin_amdgcn_wave_barrier();

    // --- O += P(16x32) * V(32x64): preload 4 V fragments, then WMMAs
    v16bf pa = cat16(*(const bf16x8*)(ps + lo * 32 + hi16),
                     *(const bf16x8*)(ps + lo * 32 + 8 + hi16));
    const bf16_t* vp = V + (size_t)lo * S + t0 + hi16;
    v16bf v0 = ldB(vp);
    v16bf v1 = ldB(vp + 16 * S);
    v16bf v2 = ldB(vp + 32 * S);
    v16bf v3 = ldB(vp + 48 * S);
    o[0] = wmma_bf16(pa, v0, o[0]);
    o[1] = wmma_bf16(pa, v1, o[1]);
    o[2] = wmma_bf16(pa, v2, o[2]);
    o[3] = wmma_bf16(pa, v3, o[3]);
    __builtin_amdgcn_wave_barrier();
  }

  // --- normalize, emit concat layout Oc[b][s][h*64 + c] (bf16)
#pragma unroll
  for (int j = 0; j < 8; ++j) {
    const float inv = 1.0f / l[j];
    const size_t base = ((size_t)b * S + s0 + j + koff) * HD + hh * DK;
#pragma unroll
    for (int g = 0; g < 4; ++g)
      Oc[base + g * 16 + lo] = (bf16_t)(o[g][j] * inv);
  }
}

// ---------------------------------------------------------------------------
// Kernel 3: output projection.  Oc(16384x512 bf16) * Wo -> f32.
// grid (M/128, N/64), block 256 (8 waves); TDM double-buffers WoT tiles.
// ---------------------------------------------------------------------------
__global__ __launch_bounds__(256) void oproj_kernel(
    const bf16_t* __restrict__ Oc, const bf16_t* __restrict__ WoT,
    float* __restrict__ out) {
  constexpr int N = 512, Kd = 512;
  const int wave = threadIdx.x >> 5, lane = threadIdx.x & 31;
  const int lo = lane & 15, hi = lane >> 4;
  const int koff = hi * 8, hi16 = hi * 16;
  const int wrow = blockIdx.x * 128 + wave * 16;
  const int colBase = blockIdx.y * 64;

  __shared__ __align__(16) bf16_t wbuf[2][64 * 32];

  const bf16_t* ap = Oc + (size_t)(wrow + lo) * Kd;
  const bf16_t* Wc = WoT + (size_t)colBase * Kd;   // [c][k], c from colBase
  v8f acc[4] = {};

#if HAVE_TDM
  if (wave == 0) tdm_load_tile_32x64((unsigned)(uintptr_t)&wbuf[0][0], Wc, 512);
#else
  stage_tile_32x64(wbuf[0], Wc, 512);
#endif

  int p = 0;
#pragma unroll 1
  for (int kk = 0; kk < Kd; kk += 32) {
    __syncthreads();
#if HAVE_TDM
    if (wave == 0) {
      if (kk + 32 < Kd)
        tdm_load_tile_32x64((unsigned)(uintptr_t)&wbuf[p ^ 1][0], Wc + kk + 32, 512);
      if (kk + 32 < Kd) __builtin_amdgcn_s_wait_tensorcnt((short)1);
      else              __builtin_amdgcn_s_wait_tensorcnt((short)0);
    }
#else
    if (kk + 32 < Kd) stage_tile_32x64(wbuf[p ^ 1], Wc + kk + 32, 512);
#endif
    __syncthreads();

    __builtin_prefetch(ap + kk + 64, 0, 0);

    v16bf a = cat16(*(const bf16x8*)(ap + kk + hi16),
                    *(const bf16x8*)(ap + kk + 8 + hi16));
    const bf16_t* wb = wbuf[p];
    v16bf b0 = ldB(wb + (0 * 16 + lo) * 32 + hi16);
    v16bf b1 = ldB(wb + (1 * 16 + lo) * 32 + hi16);
    v16bf b2 = ldB(wb + (2 * 16 + lo) * 32 + hi16);
    v16bf b3 = ldB(wb + (3 * 16 + lo) * 32 + hi16);
    acc[0] = wmma_bf16(a, b0, acc[0]);
    acc[1] = wmma_bf16(a, b1, acc[1]);
    acc[2] = wmma_bf16(a, b2, acc[2]);
    acc[3] = wmma_bf16(a, b3, acc[3]);
    p ^= 1;
  }

#pragma unroll
  for (int g = 0; g < 4; ++g)
#pragma unroll
    for (int j = 0; j < 8; ++j)
      out[(size_t)(wrow + j + koff) * N + colBase + g * 16 + lo] = acc[g][j];
}

// ---------------------------------------------------------------------------
extern "C" void kernel_launch(void* const* d_in, const int* in_sizes, int n_in,
                              void* d_out, int out_size, void* d_ws, size_t ws_size,
                              hipStream_t stream) {
  const float* q    = (const float*)d_in[0];
  const float* k    = (const float*)d_in[1];
  const float* v    = (const float*)d_in[2];
  const float* mask = (const float*)d_in[3];
  const float* Wq   = (const float*)d_in[4];
  const float* Wk   = (const float*)d_in[5];
  const float* Wv   = (const float*)d_in[6];
  const float* Wo   = (const float*)d_in[7];
  float* out = (float*)d_out;

  const size_t per = (size_t)32 * 8 * 512 * 64;   // elems per bf16 tensor
  bf16_t* Qb  = (bf16_t*)d_ws;
  bf16_t* Kb  = Qb + per;
  bf16_t* Vt  = Kb + per;
  bf16_t* Oc  = Vt + per;
  bf16_t* Wt  = Oc + per;                         // 3*8*64*512 elems
  bf16_t* WoT = Wt + (size_t)3 * 8 * 64 * 512;    // 512*512 elems

  prep_weights<<<dim3(4096), 256, 0, stream>>>(Wq, Wk, Wv, Wo, Wt, WoT);
  proj_kernel<<<dim3(4, 256, 3), 256, 0, stream>>>(q, k, v, Wt, Qb, Kb, Vt);
  attn_kernel<<<dim3(8, 256), 128, 0, stream>>>(Qb, Kb, Vt, mask, Oc);
  oproj_kernel<<<dim3(128, 8), 256, 0, stream>>>(Oc, WoT, out);
}
